// Dlrm_6116033429828
// MI455X (gfx1250) — matrix operand order repaired and checked by reading, not compiled, not run on hardware
//
#include <hip/hip_runtime.h>
#include <math.h>

typedef __bf16 bf16_t;
typedef __attribute__((ext_vector_type(16))) __bf16 v16bf;
typedef __attribute__((ext_vector_type(8)))  __bf16 v8bf;
typedef __attribute__((ext_vector_type(8)))  float  v8f;

#define BATCH     16384
#define EMB_DIM   128
#define NUM_CAT   26
#define TABLE_SZ  100000
#define NUM_FEATS 27
#define TOP_IN    480      // 128 bottom + 351 tril + 1 pad
#define ISTRIDE   136      // LDS row stride (bf16 elems): 272B/row -> 4-dword bank offset per row

static __device__ inline v8f zero8() {
  v8f z;
  #pragma unroll
  for (int i = 0; i < 8; ++i) z[i] = 0.0f;
  return z;
}

union BF16x16 { v16bf v; v8bf h[2]; };

// A-fragment (16x32 MxK, bf16), row-major source with leading dim ld.
// lane<16 : M=lane,  K(e) = k0 + {e   | e<8 ; e+8  | e>=8}
// lane>=16: M=lane-16, K(e) = k0 + {e+8 | e<8 ; e+16 | e>=8}
static __device__ inline v16bf load_a_frag(const bf16_t* X, int ld, int row, int k0, int lane) {
  int half = lane >> 4;
  int r = row + (lane & 15);
  const bf16_t* p = X + (size_t)r * ld + k0 + half * 8;
  BF16x16 f;
  f.h[0] = *(const v8bf*)(p);        // K = k0 + half*8 + [0..7]
  f.h[1] = *(const v8bf*)(p + 16);   // K = k0 + 16 + half*8 + [0..7]
  return f.v;
}

// B-fragment (32x16 KxN, bf16) where B[k][n] = W[n][k], W row-major [N x ld].
// lane&15 = n-offset; K(e) = k0 + {8*half + e | e<8 ; 16 + 8*half + (e-8) | e>=8}
static __device__ inline v16bf load_b_frag(const bf16_t* W, int ld, int n0, int k0, int lane) {
  int half = lane >> 4;
  int n = n0 + (lane & 15);
  const bf16_t* p = W + (size_t)n * ld + k0 + half * 8;
  BF16x16 f;
  f.h[0] = *(const v8bf*)(p);
  f.h[1] = *(const v8bf*)(p + 16);
  return f.v;
}

static __device__ inline v8f wmma_bf16(v16bf a, v16bf b, v8f c) {
  return __builtin_amdgcn_wmma_f32_16x16x32_bf16(false, a, false, b, (short)0, c, false, false);
}

// ---------------------------------------------------------------------------
// fp32 -> bf16 convert with optional K padding (kin <= kout; pad with zeros)
// ---------------------------------------------------------------------------
__global__ void cvt_pad_kernel(const float* __restrict__ src, bf16_t* __restrict__ dst,
                               int rows, int kin, int kout) {
  size_t i = (size_t)blockIdx.x * blockDim.x + threadIdx.x;
  size_t total = (size_t)rows * kout;
  if (i >= total) return;
  int r = (int)(i / kout);
  int c = (int)(i % kout);
  dst[i] = (c < kin) ? (bf16_t)src[(size_t)r * kin + c] : (bf16_t)0.0f;
}

// ---------------------------------------------------------------------------
// Y[M x N] = relu(X[M x K] @ W[N x K]^T + bias[N]) ; all bf16 except bias/acc
// One wave computes a 32(M) x 64(N) tile: 2 A-frags, 4 B-frags, 8 WMMAs/K-step.
// Requires M%32==0, N%64==0, K%32==0.
// ---------------------------------------------------------------------------
__global__ void gemm_relu_kernel(const bf16_t* __restrict__ X, const bf16_t* __restrict__ W,
                                 const float* __restrict__ bias, bf16_t* __restrict__ Y,
                                 int M, int N, int K) {
  int lane = threadIdx.x & 31;
  int wid  = blockIdx.x * (blockDim.x >> 5) + (threadIdx.x >> 5);
  int wpr  = N >> 6;                       // wave-tiles per M-row
  int m0 = (wid / wpr) << 5;
  int n0 = (wid % wpr) << 6;
  if (m0 >= M) return;

  v8f acc[2][4];
  #pragma unroll
  for (int i = 0; i < 2; ++i)
    #pragma unroll
    for (int j = 0; j < 4; ++j) acc[i][j] = zero8();

  for (int k0 = 0; k0 < K; k0 += 32) {
    v16bf a0 = load_a_frag(X, K, m0,      k0, lane);
    v16bf a1 = load_a_frag(X, K, m0 + 16, k0, lane);
    v16bf b0 = load_b_frag(W, K, n0,      k0, lane);
    v16bf b1 = load_b_frag(W, K, n0 + 16, k0, lane);
    v16bf b2 = load_b_frag(W, K, n0 + 32, k0, lane);
    v16bf b3 = load_b_frag(W, K, n0 + 48, k0, lane);
    acc[0][0] = wmma_bf16(a0, b0, acc[0][0]);
    acc[0][1] = wmma_bf16(a0, b1, acc[0][1]);
    acc[0][2] = wmma_bf16(a0, b2, acc[0][2]);
    acc[0][3] = wmma_bf16(a0, b3, acc[0][3]);
    acc[1][0] = wmma_bf16(a1, b0, acc[1][0]);
    acc[1][1] = wmma_bf16(a1, b1, acc[1][1]);
    acc[1][2] = wmma_bf16(a1, b2, acc[1][2]);
    acc[1][3] = wmma_bf16(a1, b3, acc[1][3]);
  }

  int half = lane >> 4;
  int nl   = lane & 15;
  #pragma unroll
  for (int i = 0; i < 2; ++i) {
    #pragma unroll
    for (int j = 0; j < 4; ++j) {
      int n = n0 + j * 16 + nl;
      float bv = bias[n];
      #pragma unroll
      for (int e = 0; e < 8; ++e) {
        int m = m0 + i * 16 + e + half * 8;     // C/D layout: M = e + 8*half
        float v = acc[i][j][e] + bv;
        v = v > 0.0f ? v : 0.0f;
        Y[(size_t)m * N + n] = (bf16_t)v;
      }
    }
  }
}

// ---------------------------------------------------------------------------
// Embedding gather + pairwise interaction + assemble top-MLP input row (480).
// One wave per sample: F (32x128 bf16, rows 27..31 zero) in LDS, G = F F^T via
// 16 WMMAs, scatter strict-lower-triangle (351) + bottom (128) + pad (1).
// ---------------------------------------------------------------------------
__global__ void interact_kernel(const bf16_t* __restrict__ BOT, const int* __restrict__ cat,
                                const float* __restrict__ emb, bf16_t* __restrict__ XT) {
  __shared__ bf16_t F[4][32][ISTRIDE];
  int lane = threadIdx.x & 31;
  int w    = threadIdx.x >> 5;
  int b    = blockIdx.x * 4 + w;
  bf16_t (*Fw)[ISTRIDE] = F[w];

  // row 0: bottom-MLP output
  #pragma unroll
  for (int c = lane; c < 128; c += 32)
    Fw[0][c] = BOT[(size_t)b * 128 + c];

  // rows 1..26: embedding gather (one float4 per lane per table)
  for (int t = 0; t < NUM_CAT; ++t) {
    int idx = cat[(size_t)t * BATCH + b];
    const float* src = emb + ((size_t)t * TABLE_SZ + (size_t)idx) * EMB_DIM;
    float4 v = *(const float4*)(src + lane * 4);
    int c = lane * 4;
    Fw[1 + t][c + 0] = (bf16_t)v.x;
    Fw[1 + t][c + 1] = (bf16_t)v.y;
    Fw[1 + t][c + 2] = (bf16_t)v.z;
    Fw[1 + t][c + 3] = (bf16_t)v.w;
  }
  // rows 27..31: zero pad
  #pragma unroll
  for (int r = 27; r < 32; ++r)
    for (int c = lane; c < 128; c += 32) Fw[r][c] = (bf16_t)0.0f;

  __syncthreads();

  v8f acc[2][2];
  #pragma unroll
  for (int i = 0; i < 2; ++i)
    #pragma unroll
    for (int j = 0; j < 2; ++j) acc[i][j] = zero8();

  const bf16_t* Fp = &Fw[0][0];
  #pragma unroll
  for (int k0 = 0; k0 < 128; k0 += 32) {
    v16bf a0 = load_a_frag(Fp, ISTRIDE, 0,  k0, lane);
    v16bf a1 = load_a_frag(Fp, ISTRIDE, 16, k0, lane);
    v16bf b0 = load_b_frag(Fp, ISTRIDE, 0,  k0, lane);  // B[k][n] = F[n][k] -> F^T
    v16bf b1 = load_b_frag(Fp, ISTRIDE, 16, k0, lane);
    acc[0][0] = wmma_bf16(a0, b0, acc[0][0]);
    acc[0][1] = wmma_bf16(a0, b1, acc[0][1]);
    acc[1][0] = wmma_bf16(a1, b0, acc[1][0]);
    acc[1][1] = wmma_bf16(a1, b1, acc[1][1]);
  }

  bf16_t* xrow = XT + (size_t)b * TOP_IN;
  #pragma unroll
  for (int c = lane; c < 128; c += 32) xrow[c] = Fw[0][c];
  if (lane == 0) xrow[TOP_IN - 1] = (bf16_t)0.0f;

  int half = lane >> 4;
  int nl   = lane & 15;
  #pragma unroll
  for (int i = 0; i < 2; ++i) {
    #pragma unroll
    for (int j = 0; j < 2; ++j) {
      int c = j * 16 + nl;
      #pragma unroll
      for (int e = 0; e < 8; ++e) {
        int r = i * 16 + e + half * 8;
        if (r < NUM_FEATS && c < r)
          xrow[128 + (r * (r - 1)) / 2 + c] = (bf16_t)acc[i][j][e];
      }
    }
  }
}

// ---------------------------------------------------------------------------
// out[b] = sigmoid(T4[b,:256] . w5 + b5)
// ---------------------------------------------------------------------------
__global__ void final_kernel(const bf16_t* __restrict__ T4, const float* __restrict__ w5,
                             const float* __restrict__ b5, float* __restrict__ out) {
  int b = blockIdx.x * blockDim.x + threadIdx.x;
  if (b >= BATCH) return;
  const bf16_t* x = T4 + (size_t)b * 256;
  float s = b5[0];
  #pragma unroll 8
  for (int i = 0; i < 256; ++i) s += (float)x[i] * w5[i];
  out[b] = 1.0f / (1.0f + __expf(-s));
}

// ---------------------------------------------------------------------------
extern "C" void kernel_launch(void* const* d_in, const int* in_sizes, int n_in,
                              void* d_out, int out_size, void* d_ws, size_t ws_size,
                              hipStream_t stream) {
  (void)in_sizes; (void)n_in; (void)out_size; (void)ws_size;
  const float* num = (const float*)d_in[0];
  const int*   cat = (const int*)d_in[1];
  const float* emb = (const float*)d_in[2];
  const float* bw1 = (const float*)d_in[3];  const float* bb1 = (const float*)d_in[4];
  const float* bw2 = (const float*)d_in[5];  const float* bb2 = (const float*)d_in[6];
  const float* bw3 = (const float*)d_in[7];  const float* bb3 = (const float*)d_in[8];
  const float* tw1 = (const float*)d_in[9];  const float* tb1 = (const float*)d_in[10];
  const float* tw2 = (const float*)d_in[11]; const float* tb2 = (const float*)d_in[12];
  const float* tw3 = (const float*)d_in[13]; const float* tb3 = (const float*)d_in[14];
  const float* tw4 = (const float*)d_in[15]; const float* tb4 = (const float*)d_in[16];
  const float* tw5 = (const float*)d_in[17]; const float* tb5 = (const float*)d_in[18];
  float* out = (float*)d_out;

  char* ws = (char*)d_ws;
  size_t off = 0;
  auto alloc = [&](size_t elems) -> bf16_t* {
    bf16_t* p = (bf16_t*)(ws + off);
    off += ((elems * sizeof(bf16_t)) + 255) & ~(size_t)255;
    return p;
  };
  bf16_t* X0  = alloc((size_t)BATCH * 32);
  bf16_t* H1  = alloc((size_t)BATCH * 512);
  bf16_t* H2  = alloc((size_t)BATCH * 256);
  bf16_t* BOT = alloc((size_t)BATCH * 128);
  bf16_t* XT  = alloc((size_t)BATCH * TOP_IN);
  bf16_t* T1  = alloc((size_t)BATCH * 1024);
  bf16_t* T2  = alloc((size_t)BATCH * 1024);
  bf16_t* T3  = alloc((size_t)BATCH * 512);
  bf16_t* T4  = alloc((size_t)BATCH * 256);
  bf16_t* Wb1 = alloc((size_t)512 * 32);
  bf16_t* Wb2 = alloc((size_t)256 * 512);
  bf16_t* Wb3 = alloc((size_t)128 * 256);
  bf16_t* Wt1 = alloc((size_t)1024 * TOP_IN);
  bf16_t* Wt2 = alloc((size_t)1024 * 1024);
  bf16_t* Wt3 = alloc((size_t)512 * 1024);
  bf16_t* Wt4 = alloc((size_t)256 * 512);

  auto cvt = [&](const float* s, bf16_t* d, int rows, int kin, int kout) {
    size_t total = (size_t)rows * kout;
    int blocks = (int)((total + 255) / 256);
    cvt_pad_kernel<<<blocks, 256, 0, stream>>>(s, d, rows, kin, kout);
  };
  cvt(num, X0,  BATCH, 13, 32);
  cvt(bw1, Wb1, 512,   13, 32);
  cvt(bw2, Wb2, 256,  512, 512);
  cvt(bw3, Wb3, 128,  256, 256);
  cvt(tw1, Wt1, 1024, TOP_IN, TOP_IN);
  cvt(tw2, Wt2, 1024, 1024, 1024);
  cvt(tw3, Wt3, 512,  1024, 1024);
  cvt(tw4, Wt4, 256,  512, 512);

  auto gemm = [&](const bf16_t* X, const bf16_t* W, const float* bias, bf16_t* Y,
                  int M, int N, int K) {
    int waves = (M / 32) * (N / 64);
    gemm_relu_kernel<<<waves / 8, 256, 0, stream>>>(X, W, bias, Y, M, N, K);
  };

  // Bottom MLP
  gemm(X0, Wb1, bb1, H1,  BATCH, 512, 32);
  gemm(H1, Wb2, bb2, H2,  BATCH, 256, 512);
  gemm(H2, Wb3, bb3, BOT, BATCH, 128, 256);

  // Embedding gather + interaction + assemble top input
  interact_kernel<<<BATCH / 4, 128, 0, stream>>>(BOT, cat, emb, XT);

  // Top MLP
  gemm(XT, Wt1, tb1, T1, BATCH, 1024, TOP_IN);
  gemm(T1, Wt2, tb2, T2, BATCH, 1024, 1024);
  gemm(T2, Wt3, tb3, T3, BATCH, 512, 1024);
  gemm(T3, Wt4, tb4, T4, BATCH, 256, 512);

  // Final 256 -> 1 + sigmoid
  final_kernel<<<BATCH / 256, 256, 0, stream>>>(T4, tw5, tb5, out);
}
